// SlotAttentionMod_58076547776893
// MI455X (gfx1250) — matrix operand order repaired and checked by reading, not compiled, not run on hardware
//
#include <hip/hip_runtime.h>
#include <math.h>

typedef __attribute__((ext_vector_type(16))) _Float16 v16h;
typedef __attribute__((ext_vector_type(8)))  _Float16 v8h;
typedef __attribute__((ext_vector_type(8)))  float    v8f;

#define CB 16
#define CN 4096
#define CNS 24
#define CD 512
#define CH 8
#define CDH 64
#define CK 512
#define CR 8
#define CITERS 3
#define CEPS 1e-8f
#define CLORA_SCALE 8.0f

// ---------------------------------------------------------------------------
// Generic NT WMMA GEMM:  C[M,N] = A[M,K] * Bt[N,K]^T   (f16 in, f32 accum)
// Each wave computes a 16(M) x 64(N) strip: one A fragment is reused across
// four B fragments per k-step (4 wmma / 10 b128 loads). 4 waves per block
// take consecutive M tiles, so a block covers 64x64 of C.
// Out-of-range A rows are CLAMPED (loads stay convergent, EXEC stays all-1s
// for WMMA); the store guard discards padded rows.
// Optional batching: blockIdx.z -> (bb, hh) with independent base strides.
// ---------------------------------------------------------------------------
__global__ __launch_bounds__(128)
void gemm_nt_kernel(const _Float16* __restrict__ A, long lda,
                    const _Float16* __restrict__ Bt, long ldb,
                    float* __restrict__ Cf, _Float16* __restrict__ Ch, long ldc,
                    int M, int Nn, int Kd,
                    int nbh, long sAb, long sAh, long sBb, long sBh,
                    long sCb, long sCh)
{
    const int z  = blockIdx.z;
    const int bb = z / nbh;
    const int hh = z % nbh;
    const _Float16* Ab = A  + (long)bb * sAb + (long)hh * sAh;
    const _Float16* Bb = Bt + (long)bb * sBb + (long)hh * sBh;
    const long cbase   = (long)bb * sCb + (long)hh * sCh;

    const int wave = threadIdx.x >> 5;
    const int lane = threadIdx.x & 31;
    const int m0   = (blockIdx.y * 4 + wave) * 16;
    const int n0   = blockIdx.x * 64;
    if (m0 >= M) return;                         // whole-wave exit, no barriers

    const int lr = lane & 15;
    const int lh = lane >> 4;
    int arow = m0 + lr;
    if (arow >= M) arow = M - 1;                 // clamp: loaded, never stored

    int brow[4];
    #pragma unroll
    for (int t = 0; t < 4; ++t) {
        int r = n0 + t * 16 + lr;
        brow[t] = (r < Nn) ? r : (Nn - 1);       // clamp (N always mult of 16)
    }

    v8f acc[4] = {};
    const _Float16* arp = Ab + (long)arow * lda + lh * 8;
    for (int k0 = 0; k0 < Kd; k0 += 32) {
        v16h a;
        {
            const v8h a0 = *(const v8h*)(arp + k0);
            const v8h a1 = *(const v8h*)(arp + k0 + 16);
            #pragma unroll
            for (int t = 0; t < 8; ++t) { a[t] = a0[t]; a[8 + t] = a1[t]; }
        }
        #pragma unroll
        for (int t = 0; t < 4; ++t) {
            v16h b;
            const _Float16* brp = Bb + (long)brow[t] * ldb + k0 + lh * 8;
            const v8h b0 = *(const v8h*)(brp);
            const v8h b1 = *(const v8h*)(brp + 16);
            #pragma unroll
            for (int u = 0; u < 8; ++u) { b[u] = b0[u]; b[8 + u] = b1[u]; }
            acc[t] = __builtin_amdgcn_wmma_f32_16x16x32_f16(false, a, false, b,
                                                            (short)0, acc[t],
                                                            false, false);
        }
    }

    #pragma unroll
    for (int t = 0; t < 4; ++t) {
        const int col = n0 + t * 16 + lr;
        if (col < Nn) {
            #pragma unroll
            for (int r = 0; r < 8; ++r) {
                const int row = m0 + lh * 8 + r;
                if (row < M) {
                    const long o = cbase + (long)row * ldc + col;
                    if (Cf) Cf[o] = acc[t][r];
                    if (Ch) Ch[o] = (_Float16)acc[t][r];
                }
            }
        }
    }
}

// ---------------------------------------------------------------------------
// LayerNorm over rows of [rows, Dc], output f16. blockDim = 256.
// ---------------------------------------------------------------------------
__global__ __launch_bounds__(256)
void ln_f16_kernel(const float* __restrict__ X, const float* __restrict__ g,
                   const float* __restrict__ bta, _Float16* __restrict__ Y, int Dc)
{
    const long row = blockIdx.x;
    const float* x = X + row * (long)Dc;
    float s = 0.f, s2 = 0.f;
    for (int d = threadIdx.x; d < Dc; d += 256) { float v = x[d]; s += v; s2 += v * v; }
    __shared__ float sm[256], sm2[256];
    sm[threadIdx.x] = s; sm2[threadIdx.x] = s2;
    __syncthreads();
    for (int o = 128; o > 0; o >>= 1) {
        if (threadIdx.x < o) { sm[threadIdx.x] += sm[threadIdx.x + o];
                               sm2[threadIdx.x] += sm2[threadIdx.x + o]; }
        __syncthreads();
    }
    const float mean = sm[0] / Dc;
    const float var  = sm2[0] / Dc - mean * mean;
    const float inv  = rsqrtf(var + 1e-5f);
    for (int d = threadIdx.x; d < Dc; d += 256)
        Y[row * (long)Dc + d] = (_Float16)((x[d] - mean) * inv * g[d] + bta[d]);
}

__global__ void cast_f16_kernel(const float* __restrict__ X, _Float16* __restrict__ Y, long n)
{
    long i = (long)blockIdx.x * blockDim.x + threadIdx.x;
    if (i < n) Y[i] = (_Float16)X[i];
}

__global__ void transpose_cast_kernel(const float* __restrict__ W, _Float16* __restrict__ WT,
                                      int Rr, int Cc)
{
    long i = (long)blockIdx.x * blockDim.x + threadIdx.x;
    if (i >= (long)Rr * Cc) return;
    int r = (int)(i / Cc), c = (int)(i % Cc);
    WT[(long)c * Rr + r] = (_Float16)W[i];
}

__global__ void copy_f32_kernel(const float* __restrict__ a, float* __restrict__ b, long n)
{
    long i = (long)blockIdx.x * blockDim.x + threadIdx.x;
    if (i < n) b[i] = a[i];
}

__global__ __launch_bounds__(256)
void rownorm_kernel(const float* __restrict__ X, float* __restrict__ out, int Dc)
{
    const long row = blockIdx.x;
    float s = 0.f;
    for (int d = threadIdx.x; d < Dc; d += 256) { float v = X[row * (long)Dc + d]; s += v * v; }
    __shared__ float sm[256];
    sm[threadIdx.x] = s; __syncthreads();
    for (int o = 128; o > 0; o >>= 1) {
        if (threadIdx.x < o) sm[threadIdx.x] += sm[threadIdx.x + o];
        __syncthreads();
    }
    if (threadIdx.x == 0) out[row] = sm[0];
}

// softmax over the (h,i) axis for each (b, j): S = H*NS = 192 strided values
__global__ __launch_bounds__(256)
void softmax_slots_kernel(float* __restrict__ p, float scale)
{
    const int b = blockIdx.y;
    const int j = blockIdx.x * 256 + threadIdx.x;
    if (j >= CN) return;
    const long base = (long)b * CH * CNS * CN + j;
    const int S = CH * CNS;
    float mx = -1e30f;
    for (int s = 0; s < S; ++s) mx = fmaxf(mx, p[base + (long)s * CN] * scale);
    float sum = 0.f;
    for (int s = 0; s < S; ++s) sum += __expf(p[base + (long)s * CN] * scale - mx);
    const float inv = 1.f / sum;
    for (int s = 0; s < S; ++s) {
        long o = base + (long)s * CN;
        p[o] = __expf(p[o] * scale - mx) * inv;
    }
}

// per (b,h,i): invrs = 1 / (sum_j p + N*EPS)
__global__ __launch_bounds__(256)
void rowsum_kernel(const float* __restrict__ p, float* __restrict__ invrs)
{
    const long base = (long)blockIdx.x * CN;
    float s = 0.f;
    for (int j = threadIdx.x; j < CN; j += 256) s += p[base + j];
    __shared__ float sm[256];
    sm[threadIdx.x] = s; __syncthreads();
    for (int o = 128; o > 0; o >>= 1) {
        if (threadIdx.x < o) sm[threadIdx.x] += sm[threadIdx.x + o];
        __syncthreads();
    }
    if (threadIdx.x == 0) invrs[blockIdx.x] = 1.f / (sm[0] + CEPS * (float)CN);
}

// upd[b,i,h,d] = invrs * sum_j (p[b,h,i,j]+EPS) * v[b,j,h,d].  block = 64 thr.
__global__ __launch_bounds__(64)
void upd_kernel(const _Float16* __restrict__ vh, const float* __restrict__ p,
                const float* __restrict__ invrs, float* __restrict__ upd)
{
    const int row = blockIdx.x;            // b*NS + i
    const int h   = blockIdx.y;
    const int b = row / CNS, i = row % CNS;
    const int d = threadIdx.x;
    const long pbase = (((long)b * CH + h) * CNS + i) * CN;
    const long vbase = (long)b * CN * CD + h * CDH + d;
    float a0 = 0.f, a1 = 0.f, a2 = 0.f, a3 = 0.f;
    for (int j = 0; j < CN; j += 4) {
        a0 += (p[pbase + j + 0] + CEPS) * (float)vh[vbase + (long)(j + 0) * CD];
        a1 += (p[pbase + j + 1] + CEPS) * (float)vh[vbase + (long)(j + 1) * CD];
        a2 += (p[pbase + j + 2] + CEPS) * (float)vh[vbase + (long)(j + 2) * CD];
        a3 += (p[pbase + j + 3] + CEPS) * (float)vh[vbase + (long)(j + 3) * CD];
    }
    upd[(long)row * CD + h * CDH + d] =
        (a0 + a1 + a2 + a3) * invrs[(b * CH + h) * CNS + i];
}

__global__ __launch_bounds__(256)
void gru_kernel(const float* __restrict__ gi, const float* __restrict__ gh,
                const float* __restrict__ bih, const float* __restrict__ bhh,
                float* __restrict__ slots)
{
    const long row = blockIdx.x;
    for (int d = threadIdx.x; d < CD; d += 256) {
        const long gb = row * 3 * CD;
        float ir  = gi[gb + d]            + bih[d];
        float iz  = gi[gb + CD + d]       + bih[CD + d];
        float inn = gi[gb + 2 * CD + d]   + bih[2 * CD + d];
        float hr  = gh[gb + d]            + bhh[d];
        float hz  = gh[gb + CD + d]       + bhh[CD + d];
        float hn  = gh[gb + 2 * CD + d]   + bhh[2 * CD + d];
        float r = 1.f / (1.f + __expf(-(ir + hr)));
        float z = 1.f / (1.f + __expf(-(iz + hz)));
        float n = tanhf(inn + r * hn);
        float h = slots[row * CD + d];
        slots[row * CD + d] = (1.f - z) * n + z * h;
    }
}

__global__ __launch_bounds__(256)
void vq_argmin_kernel(const float* __restrict__ dots, const float* __restrict__ snorm,
                      const float* __restrict__ cbn, int* __restrict__ idx)
{
    const int row = blockIdx.x;
    float best = 1e30f; int bi = 0x7fffffff;
    for (int k = threadIdx.x; k < CK; k += 256) {
        float d2 = snorm[row] - 2.f * dots[(long)row * CK + k] + cbn[k];
        if (d2 < best || (d2 == best && k < bi)) { best = d2; bi = k; }
    }
    __shared__ float bv[256]; __shared__ int bx[256];
    bv[threadIdx.x] = best; bx[threadIdx.x] = bi;
    __syncthreads();
    for (int o = 128; o > 0; o >>= 1) {
        if (threadIdx.x < o) {
            float ov = bv[threadIdx.x + o]; int oi = bx[threadIdx.x + o];
            if (ov < bv[threadIdx.x] || (ov == bv[threadIdx.x] && oi < bx[threadIdx.x])) {
                bv[threadIdx.x] = ov; bx[threadIdx.x] = oi;
            }
        }
        __syncthreads();
    }
    if (threadIdx.x == 0) idx[row] = bx[0];
}

__global__ void zerof_kernel(float* p, int n)
{
    int i = blockIdx.x * blockDim.x + threadIdx.x;
    if (i < n) p[i] = 0.f;
}

__global__ __launch_bounds__(256)
void commit_kernel(const float* __restrict__ slots, const float* __restrict__ cb,
                   const int* __restrict__ idx, float* __restrict__ commit, float invTot)
{
    const int row = blockIdx.x;
    const int id  = idx[row];
    float s = 0.f;
    for (int d = threadIdx.x; d < CD; d += 256) {
        float df = slots[(long)row * CD + d] - cb[(long)id * CD + d];
        s += df * df;
    }
    __shared__ float sm[256];
    sm[threadIdx.x] = s; __syncthreads();
    for (int o = 128; o > 0; o >>= 1) {
        if (threadIdx.x < o) sm[threadIdx.x] += sm[threadIdx.x + o];
        __syncthreads();
    }
    if (threadIdx.x == 0) atomicAdd(commit, sm[0] * invTot);
}

// Fused bias + optional rank-8 LoRA + optional relu + optional residual.
// Xg: raw GEMM output [rows,outD]; tin: f16 LoRA input [rows,inD].
__global__ __launch_bounds__(256)
void ff_kernel(const float* __restrict__ Xg, const float* __restrict__ bias,
               const _Float16* __restrict__ tin,
               const float* __restrict__ Aad, const float* __restrict__ Bad,
               const int* __restrict__ idx, int use_lora,
               const float* __restrict__ residual, int do_relu,
               float* __restrict__ Yf, _Float16* __restrict__ Yh,
               int inD, int outD, float lscale)
{
    const int row = blockIdx.x;
    __shared__ float part[CR * 256];
    __shared__ float tsh[CR];
    int id = 0;
    if (use_lora) {
        id = idx[row];
        float acc[CR];
        #pragma unroll
        for (int r = 0; r < CR; ++r) acc[r] = 0.f;
        const float* Arow = Aad + (long)id * inD * CR;
        for (int d = threadIdx.x; d < inD; d += 256) {
            float xv = (float)tin[(long)row * inD + d];
            #pragma unroll
            for (int r = 0; r < CR; ++r) acc[r] += xv * Arow[(long)d * CR + r];
        }
        #pragma unroll
        for (int r = 0; r < CR; ++r) part[r * 256 + threadIdx.x] = acc[r];
        __syncthreads();
        for (int o = 128; o > 0; o >>= 1) {
            if (threadIdx.x < o) {
                #pragma unroll
                for (int r = 0; r < CR; ++r)
                    part[r * 256 + threadIdx.x] += part[r * 256 + threadIdx.x + o];
            }
            __syncthreads();
        }
        if (threadIdx.x < CR) tsh[threadIdx.x] = part[threadIdx.x * 256];
        __syncthreads();
    }
    const float* Brow = Bad + (long)id * CR * outD;
    for (int c = threadIdx.x; c < outD; c += 256) {
        float val = Xg[(long)row * outD + c] + bias[c];
        if (use_lora) {
            float l = 0.f;
            #pragma unroll
            for (int r = 0; r < CR; ++r) l += tsh[r] * Brow[(long)r * outD + c];
            val += lscale * l;
        }
        if (do_relu) val = fmaxf(val, 0.f);
        if (residual) val += residual[(long)row * outD + c];
        Yf[(long)row * outD + c] = val;
        if (Yh) Yh[(long)row * outD + c] = (_Float16)val;
    }
}

__global__ __launch_bounds__(256)
void attn_mean_kernel(const float* __restrict__ p, float* __restrict__ attn)
{
    const int row = blockIdx.y;                 // b*NS + i
    const int j = blockIdx.x * 256 + threadIdx.x;
    if (j >= CN) return;
    const int b = row / CNS, i = row % CNS;
    float s = 0.f;
    for (int h = 0; h < CH; ++h)
        s += p[(((long)b * CH + h) * CNS + i) * CN + j];
    attn[(long)row * CN + j] = s * (1.f / CH);
}

__global__ void finalize_kernel(const int* __restrict__ idx, const float* __restrict__ commit,
                                float* __restrict__ oidx1, float* __restrict__ ocommit,
                                float* __restrict__ oidx2, int n)
{
    int t = blockIdx.x * blockDim.x + threadIdx.x;
    if (t < n) { float v = (float)idx[t]; oidx1[t] = v; oidx2[t] = v; }
    if (t == 0) ocommit[0] = commit[0];
}

// ---------------------------------------------------------------------------
extern "C" void kernel_launch(void* const* d_in, const int* in_sizes, int n_in,
                              void* d_out, int out_size, void* d_ws, size_t ws_size,
                              hipStream_t stream)
{
    (void)in_sizes; (void)n_in; (void)out_size; (void)ws_size;

    const float* inputs = (const float*)d_in[0];
    const float* cond   = (const float*)d_in[1];
    const float* Wq     = (const float*)d_in[2];
    const float* Wk     = (const float*)d_in[3];
    const float* Wv     = (const float*)d_in[4];
    const float* w_ih   = (const float*)d_in[5];
    const float* w_hh   = (const float*)d_in[6];
    const float* b_ih   = (const float*)d_in[7];
    const float* b_hh   = (const float*)d_in[8];
    const float* lnin_g = (const float*)d_in[9];
    const float* lnin_b = (const float*)d_in[10];
    const float* lns_g  = (const float*)d_in[11];
    const float* lns_b  = (const float*)d_in[12];
    const float* lnf_g  = (const float*)d_in[13];
    const float* lnf_b  = (const float*)d_in[14];
    const float* W1     = (const float*)d_in[15];
    const float* b1     = (const float*)d_in[16];
    const float* W2     = (const float*)d_in[17];
    const float* b2     = (const float*)d_in[18];
    const float* A1     = (const float*)d_in[19];
    const float* B1m    = (const float*)d_in[20];
    const float* A2     = (const float*)d_in[21];
    const float* B2m    = (const float*)d_in[22];
    const float* cb     = (const float*)d_in[23];

    // ---- workspace carve-out -------------------------------------------------
    char*  w   = (char*)d_ws;
    size_t off = 0;
    auto alloc = [&](size_t bytes) -> void* {
        void* p = w + off;
        off = (off + bytes + 255) & ~(size_t)255;
        return p;
    };
    const size_t BN = (size_t)CB * CN;     // 65536
    const size_t RS = (size_t)CB * CNS;    // 384 slot rows

    _Float16* x_h    = (_Float16*)alloc(BN * CD * 2);
    _Float16* k_h    = (_Float16*)alloc(BN * CD * 2);
    _Float16* v_h    = (_Float16*)alloc(BN * CD * 2);
    float*    p_buf  = (float*)   alloc((size_t)CB * CH * CNS * CN * 4);
    _Float16* WqT    = (_Float16*)alloc((size_t)CD * CD * 2);
    _Float16* WkT    = (_Float16*)alloc((size_t)CD * CD * 2);
    _Float16* WvT    = (_Float16*)alloc((size_t)CD * CD * 2);
    _Float16* W1T    = (_Float16*)alloc((size_t)4 * CD * CD * 2);   // [2048,512]
    _Float16* W2T    = (_Float16*)alloc((size_t)CD * 4 * CD * 2);   // [512,2048]
    _Float16* wih_h  = (_Float16*)alloc((size_t)3 * CD * CD * 2);
    _Float16* whh_h  = (_Float16*)alloc((size_t)3 * CD * CD * 2);
    _Float16* cb_h   = (_Float16*)alloc((size_t)CK * CD * 2);
    float*    cbn    = (float*)   alloc((size_t)CK * 4);
    float*    slots  = (float*)   alloc(RS * CD * 4);
    _Float16* slotsh = (_Float16*)alloc(RS * CD * 2);
    _Float16* sln_h  = (_Float16*)alloc(RS * CD * 2);
    _Float16* q_h    = (_Float16*)alloc(RS * CD * 2);
    _Float16* s2_h   = (_Float16*)alloc(RS * CD * 2);
    float*    invrs  = (float*)   alloc((size_t)CB * CH * CNS * 4);
    float*    upd    = (float*)   alloc(RS * CD * 4);
    _Float16* upd_h  = (_Float16*)alloc(RS * CD * 2);
    float*    gi     = (float*)   alloc(RS * 3 * CD * 4);
    float*    gh     = (float*)   alloc(RS * 3 * CD * 4);
    float*    h1f    = (float*)   alloc(RS * 4 * CD * 4);
    _Float16* h1_h   = (_Float16*)alloc(RS * 4 * CD * 2);
    float*    outf   = (float*)   alloc(RS * CD * 4);
    float*    vqd    = (float*)   alloc(RS * CK * 4);
    float*    snorm  = (float*)   alloc(RS * 4);
    int*      idxb   = (int*)     alloc(RS * 4);
    float*    commitb= (float*)   alloc(256);

    float* out_slots  = (float*)d_out;                               // 196608
    float* out_attn   = out_slots + RS * CD;                         // 1572864
    float* out_idx1   = out_attn + RS * CN;                          // 384
    float* out_commit = out_idx1 + RS;                               // 1
    float* out_idx2   = out_commit + 1;                              // 384

    auto gemm = [&](const _Float16* A, long lda, const _Float16* Bt, long ldb,
                    float* Cf, _Float16* Ch, long ldc, int M, int Nn, int Kd,
                    int nbatch, int nbh, long sAb, long sAh, long sBb, long sBh,
                    long sCb, long sCh) {
        dim3 g((Nn + 63) / 64, (M + 63) / 64, nbatch);
        gemm_nt_kernel<<<g, 128, 0, stream>>>(A, lda, Bt, ldb, Cf, Ch, ldc,
                                              M, Nn, Kd, nbh, sAb, sAh,
                                              sBb, sBh, sCb, sCh);
    };

    // ---- one-time prep -------------------------------------------------------
    ln_f16_kernel<<<dim3((unsigned)BN), 256, 0, stream>>>(inputs, lnin_g, lnin_b, x_h, CD);
    {
        long n = (long)CD * CD;
        transpose_cast_kernel<<<(unsigned)((n + 255) / 256), 256, 0, stream>>>(Wq, WqT, CD, CD);
        transpose_cast_kernel<<<(unsigned)((n + 255) / 256), 256, 0, stream>>>(Wk, WkT, CD, CD);
        transpose_cast_kernel<<<(unsigned)((n + 255) / 256), 256, 0, stream>>>(Wv, WvT, CD, CD);
        long n1 = (long)CD * 4 * CD;
        transpose_cast_kernel<<<(unsigned)((n1 + 255) / 256), 256, 0, stream>>>(W1, W1T, CD, 4 * CD);
        transpose_cast_kernel<<<(unsigned)((n1 + 255) / 256), 256, 0, stream>>>(W2, W2T, 4 * CD, CD);
        long ng = (long)3 * CD * CD;
        cast_f16_kernel<<<(unsigned)((ng + 255) / 256), 256, 0, stream>>>(w_ih, wih_h, ng);
        cast_f16_kernel<<<(unsigned)((ng + 255) / 256), 256, 0, stream>>>(w_hh, whh_h, ng);
        long nc = (long)CK * CD;
        cast_f16_kernel<<<(unsigned)((nc + 255) / 256), 256, 0, stream>>>(cb, cb_h, nc);
        rownorm_kernel<<<CK, 256, 0, stream>>>(cb, cbn, CD);
    }
    // k = LN(x) @ Wk ; v = LN(x) @ Wv   (f16 outputs)
    gemm(x_h, CD, WkT, CD, nullptr, k_h, CD, (int)BN, CD, CD, 1, 1, 0, 0, 0, 0, 0, 0);
    gemm(x_h, CD, WvT, CD, nullptr, v_h, CD, (int)BN, CD, CD, 1, 1, 0, 0, 0, 0, 0, 0);
    // slots = conditioning
    copy_f32_kernel<<<(unsigned)((RS * CD + 255) / 256), 256, 0, stream>>>(cond, slots, (long)(RS * CD));
    zerof_kernel<<<1, 32, 0, stream>>>(commitb, 1);

    // ---- iterations ----------------------------------------------------------
    for (int it = 0; it < CITERS; ++it) {
        // cast slots (pre-update) for VQ + GRU-hh GEMMs
        cast_f16_kernel<<<(unsigned)((RS * CD + 255) / 256), 256, 0, stream>>>(
            slots, slotsh, (long)(RS * CD));

        if (it > 0) {
            rownorm_kernel<<<(unsigned)RS, 256, 0, stream>>>(slots, snorm, CD);
            gemm(slotsh, CD, cb_h, CD, vqd, nullptr, CK, (int)RS, CK, CD,
                 1, 1, 0, 0, 0, 0, 0, 0);
            vq_argmin_kernel<<<(unsigned)RS, 256, 0, stream>>>(vqd, snorm, cbn, idxb);
            zerof_kernel<<<1, 32, 0, stream>>>(commitb, 1);
            commit_kernel<<<(unsigned)RS, 256, 0, stream>>>(
                slots, cb, idxb, commitb, 1.f / (float)(RS * CD));
        }

        // q = LN_s(slots) @ Wq  (f16)
        ln_f16_kernel<<<(unsigned)RS, 256, 0, stream>>>(slots, lns_g, lns_b, sln_h, CD);
        gemm(sln_h, CD, WqT, CD, nullptr, q_h, CD, (int)RS, CD, CD, 1, 1, 0, 0, 0, 0, 0, 0);

        // dots[b,h,i,j] = q . k   (batched over b,h)
        gemm(q_h, CD, k_h, CD, p_buf, nullptr, CN, CNS, CN, CDH,
             CB * CH, CH,
             (long)CNS * CD, (long)CDH,            // A strides (b,h)
             (long)CN * CD, (long)CDH,             // B strides (b,h)
             (long)CH * CNS * CN, (long)CNS * CN); // C strides (b,h)

        softmax_slots_kernel<<<dim3(CN / 256, CB), 256, 0, stream>>>(
            p_buf, 1.f / sqrtf((float)CDH));
        if (it == CITERS - 1)
            attn_mean_kernel<<<dim3(CN / 256, (unsigned)RS), 256, 0, stream>>>(p_buf, out_attn);

        rowsum_kernel<<<(unsigned)(CB * CH * CNS), 256, 0, stream>>>(p_buf, invrs);
        upd_kernel<<<dim3((unsigned)RS, CH), 64, 0, stream>>>(v_h, p_buf, invrs, upd);

        // GRU
        cast_f16_kernel<<<(unsigned)((RS * CD + 255) / 256), 256, 0, stream>>>(
            upd, upd_h, (long)(RS * CD));
        gemm(upd_h,  CD, wih_h, CD, gi, nullptr, 3 * CD, (int)RS, 3 * CD, CD,
             1, 1, 0, 0, 0, 0, 0, 0);
        gemm(slotsh, CD, whh_h, CD, gh, nullptr, 3 * CD, (int)RS, 3 * CD, CD,
             1, 1, 0, 0, 0, 0, 0, 0);
        gru_kernel<<<(unsigned)RS, 256, 0, stream>>>(gi, gh, b_ih, b_hh, slots);

        // FFN with residual (+LoRA when idx available)
        ln_f16_kernel<<<(unsigned)RS, 256, 0, stream>>>(slots, lnf_g, lnf_b, s2_h, CD);
        gemm(s2_h, CD, W1T, CD, h1f, nullptr, 4 * CD, (int)RS, 4 * CD, CD,
             1, 1, 0, 0, 0, 0, 0, 0);
        ff_kernel<<<(unsigned)RS, 256, 0, stream>>>(
            h1f, b1, s2_h, A1, B1m, idxb, (it > 0) ? 1 : 0,
            nullptr, 1, h1f, h1_h, CD, 4 * CD, CLORA_SCALE / (float)CR);
        gemm(h1_h, 4 * CD, W2T, 4 * CD, outf, nullptr, CD, (int)RS, CD, 4 * CD,
             1, 1, 0, 0, 0, 0, 0, 0);
        ff_kernel<<<(unsigned)RS, 256, 0, stream>>>(
            outf, b2, h1_h, A2, B2m, idxb, (it > 0) ? 1 : 0,
            slots, 0, slots, nullptr, 4 * CD, CD, CLORA_SCALE / (float)CR);
    }

    // ---- outputs -------------------------------------------------------------
    copy_f32_kernel<<<(unsigned)((RS * CD + 255) / 256), 256, 0, stream>>>(
        slots, out_slots, (long)(RS * CD));
    finalize_kernel<<<2, 256, 0, stream>>>(idxb, commitb, out_idx1, out_commit,
                                           out_idx2, (int)RS);
}
